// Dual_Quantize4_43645457662417
// MI455X (gfx1250) — matrix operand order repaired and checked by reading, not compiled, not run on hardware
//
#include <hip/hip_runtime.h>
#include <hip/hip_bf16.h>

// ---------------------------------------------------------------------------
// Problem geometry (from setup_inputs): B=16,H=32,W=32,D=256,K=1024
// N = B*H*W = 16384 rows per source; two sources (hr, lr); one codebook.
// ---------------------------------------------------------------------------
#define NROWS 16384
#define DDIM  256
#define KDIM  1024

typedef __attribute__((ext_vector_type(16))) __bf16          v16bf;
typedef __attribute__((ext_vector_type(8)))  float           v8f;
typedef __attribute__((ext_vector_type(8)))  unsigned short  u16x8;
typedef __attribute__((ext_vector_type(16))) unsigned short  u16x16;

__device__ __forceinline__ unsigned short f2bf(float f) {
    unsigned int u = __builtin_bit_cast(unsigned int, f);
    unsigned int r = u + 0x7fffu + ((u >> 16) & 1u);   // round-to-nearest-even
    return (unsigned short)(r >> 16);
}

// ---------------------------------------------------------------------------
// prep_x: one block (256 thr) per row. f32 -> bf16 convert + ||x||^2 per row.
// ---------------------------------------------------------------------------
__global__ __launch_bounds__(256) void prep_x(const float* __restrict__ X,
                                              unsigned short* __restrict__ Xbf,
                                              float* __restrict__ xnorm) {
    const int row = blockIdx.x;
    const int d   = threadIdx.x;
    const float x = X[row * DDIM + d];
    Xbf[row * DDIM + d] = f2bf(x);

    __shared__ float s[256];
    s[d] = x * x;
    __syncthreads();
    for (int o = 128; o > 0; o >>= 1) {
        if (d < o) s[d] += s[d + o];
        __syncthreads();
    }
    if (d == 0) xnorm[row] = s[0];
}

// ---------------------------------------------------------------------------
// prep_e: build bf16 E^T [K, D] from f32 E [D, K] (coalesced reads over k).
// ---------------------------------------------------------------------------
__global__ __launch_bounds__(256) void prep_e(const float* __restrict__ E,
                                              unsigned short* __restrict__ Et) {
    const int idx = blockIdx.x * 256 + threadIdx.x;   // 1024 blocks -> 256K elems
    const int d = idx >> 10;
    const int k = idx & (KDIM - 1);
    Et[k * DDIM + d] = f2bf(E[idx]);
}

// enorm_k: ||e_k||^2 per column k (4 blocks x 256 threads, coalesced over k).
__global__ __launch_bounds__(256) void enorm_k(const float* __restrict__ E,
                                               float* __restrict__ enorm) {
    const int k = blockIdx.x * 256 + threadIdx.x;
    float s = 0.f;
    for (int d = 0; d < DDIM; ++d) {
        const float e = E[d * KDIM + k];
        s += e * e;
    }
    enorm[k] = s;
}

// ---------------------------------------------------------------------------
// vq_gemm: dist = ||x||^2 - 2 X.E + ||e||^2, stored twice, fused argmin.
//
// One workgroup (8 waves = 256 thr) per 16-row block.
//  - 8 KB A block staged to LDS once via global_load_async_to_lds_b128
//    (ASYNCcnt path), then all 8 A fragments hoisted into VGPRs.
//  - Each wave owns 8 column tiles of 16; inner loop = one 32 B B-load +
//    one v_wmma_f32_16x16x32_bf16, K-loop in steps of 32 over D=256.
// ---------------------------------------------------------------------------
__global__ __launch_bounds__(256) void vq_gemm(const unsigned short* __restrict__ Xbf,
                                               const unsigned short* __restrict__ Et,
                                               const float* __restrict__ xnorm,
                                               const float* __restrict__ enorm,
                                               float* __restrict__ dist0,
                                               float* __restrict__ dist1,
                                               float* __restrict__ indf0,
                                               float* __restrict__ indf1,
                                               int*   __restrict__ ind_ws) {
    const int rb   = blockIdx.x;            // 16-row block
    const int wave = threadIdx.x >> 5;      // 0..7
    const int lane = threadIdx.x & 31;
    const int hl   = lane >> 4;             // half-wave select
    const int mrow = lane & 15;             // row (A) / col (B,C,D) within tile

    __shared__ unsigned short sA[16 * DDIM];   // 8 KB staged A block
    __shared__ float ldsv[8][16];
    __shared__ int   ldsi[8][16];

    // ---- async global -> LDS stage of the 16 x 256 bf16 A block -----------
    {
        const unsigned lbase = (unsigned)(uintptr_t)(&sA[0]);
        const unsigned long long gbase =
            (unsigned long long)(uintptr_t)(Xbf + (size_t)rb * 16 * DDIM);
#pragma unroll
        for (int i = 0; i < 2; ++i) {
            const int chunk = threadIdx.x + i * 256;          // 512 x 16B
            const unsigned           laddr = lbase + chunk * 16;
            const unsigned long long gaddr = gbase + (unsigned long long)chunk * 16;
            asm volatile("global_load_async_to_lds_b128 %0, %1, off"
                         :: "v"(laddr), "v"(gaddr)
                         : "memory");
        }
        asm volatile("s_wait_asynccnt 0x0" ::: "memory");
        __syncthreads();
    }

    // ---- hoist all 8 A fragments (K = 0..255) into registers --------------
    // 16-bit 16x32 A layout: half 0 -> K {d..d+7, d+16..d+23},
    //                        half 1 -> K {d+8..d+15, d+24..d+31}
    v16bf afrag[8];
#pragma unroll
    for (int ds = 0; ds < 8; ++ds) {
        const int d = ds * 32;
        u16x8 a0 = *(const u16x8*)(&sA[mrow * DDIM + d + hl * 8]);
        u16x8 a1 = *(const u16x8*)(&sA[mrow * DDIM + d + 16 + hl * 8]);
        afrag[ds] = __builtin_bit_cast(v16bf, __builtin_shufflevector(a0, a1,
            0, 1, 2, 3, 4, 5, 6, 7, 8, 9, 10, 11, 12, 13, 14, 15));
    }

    // cached ||x||^2 for the 8 rows this lane's C fragment covers
    float xn[8];
#pragma unroll
    for (int v = 0; v < 8; ++v) xn[v] = xnorm[rb * 16 + hl * 8 + v];

    float minv[8];
    int   mini[8];
#pragma unroll
    for (int v = 0; v < 8; ++v) { minv[v] = 3.4e38f; mini[v] = 0x7fffffff; }

    for (int t = 0; t < 8; ++t) {
        const int n0 = (wave * 8 + t) * 16;               // tile's first column
        const unsigned short* bcol = Et + (size_t)(n0 + mrow) * DDIM;

        v8f acc = {0.f, 0.f, 0.f, 0.f, 0.f, 0.f, 0.f, 0.f};

#pragma unroll
        for (int ds = 0; ds < 8; ++ds) {
            const int d = ds * 32;
            // B fragment, 32x16: half 0 -> K {d..d+7, d+16..d+23}, half 1 -> K {d+8..d+15, d+24..d+31}
            u16x8 b0 = *(const u16x8*)(bcol + d + hl * 8);
            u16x8 b1 = *(const u16x8*)(bcol + d + 16 + hl * 8);
            u16x16 bu = __builtin_shufflevector(b0, b1,
                0, 1, 2, 3, 4, 5, 6, 7, 8, 9, 10, 11, 12, 13, 14, 15);
            if (d + 32 < DDIM)
                __builtin_prefetch((const void*)(bcol + d + 32 + hl * 16), 0, 0);
            v16bf b = __builtin_bit_cast(v16bf, bu);
            acc = __builtin_amdgcn_wmma_f32_16x16x32_bf16(
                false, afrag[ds], false, b, (short)0, acc, false, false);
        }

        const int   col = n0 + mrow;
        const float en  = enorm[col];
#pragma unroll
        for (int v = 0; v < 8; ++v) {
            const int row  = rb * 16 + hl * 8 + v;        // C layout: VGPR v
            const float dv = xn[v] - 2.0f * acc[v] + en;
            const size_t o = (size_t)row * KDIM + col;
            dist0[o] = dv;
            dist1[o] = dv;
            if (dv < minv[v] || (dv == minv[v] && col < mini[v])) {
                minv[v] = dv;
                mini[v] = col;
            }
        }
    }

    // argmin across the 16 lanes of each half (xor masks stay within halves)
#pragma unroll
    for (int v = 0; v < 8; ++v) {
        float mv = minv[v];
        int   mi = mini[v];
        for (int m = 1; m < 16; m <<= 1) {
            const float ov = __shfl_xor(mv, m, 32);
            const int   oi = __shfl_xor(mi, m, 32);
            if (ov < mv || (ov == mv && oi < mi)) { mv = ov; mi = oi; }
        }
        if (mrow == 0) {
            ldsv[wave][hl * 8 + v] = mv;
            ldsi[wave][hl * 8 + v] = mi;
        }
    }
    __syncthreads();

    // combine the 8 waves' per-row candidates
    if (threadIdx.x < 16) {
        const int rl = threadIdx.x;
        float bv = ldsv[0][rl];
        int   bi = ldsi[0][rl];
        for (int w = 1; w < 8; ++w) {
            const float ov = ldsv[w][rl];
            const int   oi = ldsi[w][rl];
            if (ov < bv || (ov == bv && oi < bi)) { bv = ov; bi = oi; }
        }
        const int row = rb * 16 + rl;
        ind_ws[row] = bi;
        const float fi = (float)bi;
        indf0[row] = fi;
        indf1[row] = fi;
    }
}

// ---------------------------------------------------------------------------
// gather_k: quant = E^T[ind] (from the ORIGINAL f32 codebook), write to both
// qst slots, and per-row partial sum of (q - x)^2 (deterministic tree).
// ---------------------------------------------------------------------------
__global__ __launch_bounds__(256) void gather_k(const float* __restrict__ X,
                                                const float* __restrict__ E,
                                                const int* __restrict__ ind,
                                                float* __restrict__ q0,
                                                float* __restrict__ q1,
                                                float* __restrict__ partial) {
    const int row = blockIdx.x;
    const int d   = threadIdx.x;
    const int k   = ind[row];
    const float e = E[d * KDIM + k];
    const float x = X[row * DDIM + d];
    q0[row * DDIM + d] = e;
    q1[row * DDIM + d] = e;

    __shared__ float s[256];
    const float df = e - x;
    s[d] = df * df;
    __syncthreads();
    for (int o = 128; o > 0; o >>= 1) {
        if (d < o) s[d] += s[d + o];
        __syncthreads();
    }
    if (d == 0) partial[row] = s[0];
}

// finalize: deterministic reduction of 16384 partials -> mean, two slots.
__global__ __launch_bounds__(256) void finalize(const float* __restrict__ partial,
                                                float* __restrict__ out0,
                                                float* __restrict__ out1) {
    __shared__ float s[256];
    const int t = threadIdx.x;
    float acc = 0.f;
    for (int i = t; i < NROWS; i += 256) acc += partial[i];
    s[t] = acc;
    __syncthreads();
    for (int o = 128; o > 0; o >>= 1) {
        if (t < o) s[t] += s[t + o];
        __syncthreads();
    }
    if (t == 0) {
        const float m = s[0] / ((float)NROWS * (float)DDIM);
        *out0 = m;
        *out1 = m;
    }
}

// ---------------------------------------------------------------------------
extern "C" void kernel_launch(void* const* d_in, const int* in_sizes, int n_in,
                              void* d_out, int out_size, void* d_ws, size_t ws_size,
                              hipStream_t stream) {
    (void)in_sizes; (void)n_in; (void)out_size; (void)ws_size;

    const float* in_hr = (const float*)d_in[0];
    const float* in_lr = (const float*)d_in[1];
    const float* embed = (const float*)d_in[2];   // embed_lr (embed_hr unused)
    float* out = (float*)d_out;
    char*  ws  = (char*)d_ws;

    // ---- workspace layout (~17.7 MB) ----
    unsigned short* Xhr  = (unsigned short*)(ws + 0);
    unsigned short* Xlr  = (unsigned short*)(ws + 8388608);
    unsigned short* Et   = (unsigned short*)(ws + 16777216);
    float* xnhr = (float*)(ws + 17301504);
    float* xnlr = (float*)(ws + 17367040);
    float* en   = (float*)(ws + 17432576);
    int*   ihr  = (int*)  (ws + 17436672);
    int*   ilr  = (int*)  (ws + 17502208);
    float* phr  = (float*)(ws + 17567744);
    float* plr  = (float*)(ws + 17633280);

    // ---- output layout (flat concat of the 16-tuple) ----
    const size_t QS = (size_t)NROWS * DDIM;   // 4,194,304
    const size_t DS = (size_t)NROWS * KDIM;   // 16,777,216
    const size_t IN = (size_t)NROWS;          // 16,384
    float* q_hr0 = out;
    float* q_lr0 = out + QS;
    float* q_hr1 = out + 2 * QS;
    float* q_lr1 = out + 3 * QS;
    float* diffs = out + 4 * QS;              // [hr, lr, hr, lr]
    float* if_hr0 = diffs + 4;
    float* if_lr0 = if_hr0 + IN;
    float* if_hr1 = if_hr0 + 2 * IN;
    float* if_lr1 = if_hr0 + 3 * IN;
    float* d_hr0  = if_hr0 + 4 * IN;
    float* d_lr0  = d_hr0 + DS;
    float* d_hr1  = d_hr0 + 2 * DS;
    float* d_lr1  = d_hr0 + 3 * DS;

    prep_x<<<NROWS, 256, 0, stream>>>(in_hr, Xhr, xnhr);
    prep_x<<<NROWS, 256, 0, stream>>>(in_lr, Xlr, xnlr);
    prep_e<<<(DDIM * KDIM) / 256, 256, 0, stream>>>(embed, Et);
    enorm_k<<<KDIM / 256, 256, 0, stream>>>(embed, en);

    vq_gemm<<<NROWS / 16, 256, 0, stream>>>(Xhr, Et, xnhr, en,
                                            d_hr0, d_hr1, if_hr0, if_hr1, ihr);
    vq_gemm<<<NROWS / 16, 256, 0, stream>>>(Xlr, Et, xnlr, en,
                                            d_lr0, d_lr1, if_lr0, if_lr1, ilr);

    gather_k<<<NROWS, 256, 0, stream>>>(in_hr, embed, ihr, q_hr0, q_hr1, phr);
    gather_k<<<NROWS, 256, 0, stream>>>(in_lr, embed, ilr, q_lr0, q_lr1, plr);

    finalize<<<1, 256, 0, stream>>>(phr, &diffs[0], &diffs[2]);
    finalize<<<1, 256, 0, stream>>>(plr, &diffs[1], &diffs[3]);
}
